// HGT_42863773614852
// MI455X (gfx1250) — matrix operand order repaired and checked by reading, not compiled, not run on hardware
//
#include <hip/hip_runtime.h>
#include <cstddef>
#include <cstdint>

#define NS 100000
#define NF 50000
#define TT 16
#define CC 128
#define HH 8
#define DD 16
#define LL 2
#define EE 200000

typedef __attribute__((ext_vector_type(16))) _Float16 v16h;
typedef __attribute__((ext_vector_type(8)))  float    v8f;

// ---------------- helpers: monotone float<->uint for atomicMax softmax ----
__device__ __forceinline__ unsigned fflip(float f) {
    unsigned u = __float_as_uint(f);
    return u ^ ((unsigned)((int)u >> 31) | 0x80000000u);
}
__device__ __forceinline__ float funflip(unsigned v) {
    unsigned u = (v & 0x80000000u) ? (v ^ 0x80000000u) : ~v;
    return __uint_as_float(u);
}
// gelu (jax tanh approx) with tanh expressed via hardware v_exp_f32:
// tanh(y) = 1 - 2/(exp(2y)+1)
__device__ __forceinline__ float gelu_f(float x) {
    float y = 0.7978845608028654f * (x + 0.044715f * x * x * x);
    float t = 1.f - 2.f / (__expf(2.f * y) + 1.f);
    return 0.5f * x * (1.f + t);
}

// ---------------- fill ----------------------------------------------------
__global__ void fill_u32_kernel(unsigned* __restrict__ p, unsigned v, long long n) {
    long long i = (long long)blockIdx.x * blockDim.x + threadIdx.x;
    if (i < n) p[i] = v;
}

// fill maxbuf with flip(-inf) and sumbuf with 0 in a single launch
__global__ void init_softmax_kernel(unsigned* __restrict__ maxb, float* __restrict__ sumb,
                                    long long n) {
    long long i = (long long)blockIdx.x * blockDim.x + threadIdx.x;
    if (i < n) { maxb[i] = 0x007FFFFFu; sumb[i] = 0.f; }
}

// ---------------- encoder: mean-pool token embeddings + relu -------------
__global__ void encoder_kernel(const int* __restrict__ tok, const float* __restrict__ emb,
                               float* __restrict__ out) {
    __shared__ int toks[TT];
    int node = blockIdx.x;
    int c = threadIdx.x;                       // 128 threads
    if (c < TT) toks[c] = tok[(size_t)node * TT + c];
    __syncthreads();
    float acc = 0.f;
#pragma unroll
    for (int i = 0; i < TT; ++i) acc += emb[(size_t)toks[i] * CC + c];
    acc *= (1.f / (float)TT);
    out[(size_t)node * CC + c] = fmaxf(acc, 0.f);
}

// ---- fold per-head a_rel/m_rel (blockdiag) into 128x128 effective W/b ----
// weff[c, h*16+e] = sum_d w[c, h*16+d] * rel[h, d, e]
__global__ void make_eff_kernel(const float* __restrict__ w, const float* __restrict__ b,
                                const float* __restrict__ rel,
                                float* __restrict__ weff, float* __restrict__ beff) {
    int idx = blockIdx.x * blockDim.x + threadIdx.x;
    if (idx >= CC * CC) return;
    int c = idx >> 7, he = idx & 127;
    int h = he >> 4, e = he & 15;
    float acc = 0.f;
#pragma unroll
    for (int d = 0; d < DD; ++d)
        acc += w[(size_t)c * CC + h * DD + d] * rel[h * 256 + d * DD + e];
    weff[idx] = acc;
    if (c == 0) {
        float bb = 0.f;
#pragma unroll
        for (int d = 0; d < DD; ++d)
            bb += b[h * DD + d] * rel[h * 256 + d * DD + e];
        beff[he] = bb;
    }
}

// ---------------- WMMA GEMM: [N,128] x [128,128] + bias, fused epilogue ---
// agelu: apply gelu to A elements while loading (fuses the activation pass)
// mode 0: out = act(A)@W + b
// mode 1: out = relu(act(A)@W + b)
// mode 2: out = g*(act(A)@W + b) + (1-g)*xold,  g = sigmoid(*skip_ptr)
__global__ __launch_bounds__(256)
void gemm_wmma_kernel(const float* __restrict__ A, const float* __restrict__ W,
                      const float* __restrict__ bias, float* __restrict__ out,
                      int n_rows, int mode, int agelu,
                      const float* __restrict__ xold, const float* __restrict__ skip_ptr) {
    __shared__ _Float16 ldsW[CC * CC];   // W transposed: ldsW[n*128 + k], 32KB
    __shared__ float    ldsB[CC];

    // stage weights (fp32 -> f16, transpose) cooperatively
    for (int idx = threadIdx.x; idx < CC * CC; idx += 256) {
        int k = idx >> 7, n = idx & 127;            // coalesced over n
        ldsW[n * CC + k] = (_Float16)W[idx];
    }
    if (threadIdx.x < CC) ldsB[threadIdx.x] = bias[threadIdx.x];
    __syncthreads();

    int wave = threadIdx.x >> 5;
    int lane = threadIdx.x & 31;
    int row0 = (blockIdx.x * 8 + wave) * 16;        // 16-row strip per wave
    if (row0 >= n_rows) return;                     // n_rows always mult of 16

    int hlf   = lane >> 4;                          // 0 or 1
    int m     = lane & 15;
    int khalf = hlf * 16;                           // A-layout K offset

    // ---- batch ALL A loads for the strip: one big clause, staggered waits
    const float* arow = A + (size_t)(row0 + m) * CC + khalf;
    float4 ar[16];
#pragma unroll
    for (int s = 0; s < 4; ++s) {
        const float4* ap4 = (const float4*)(arow + s * 32);
        ar[s * 4 + 0] = ap4[0];                     // K = khalf   .. khalf+7
        ar[s * 4 + 1] = ap4[1];
        ar[s * 4 + 2] = ap4[2];                     // K = khalf+8 .. khalf+15
        ar[s * 4 + 3] = ap4[3];
    }
    if (agelu) {                                    // fused activation on A
#pragma unroll
        for (int j = 0; j < 16; ++j) {
            ar[j].x = gelu_f(ar[j].x); ar[j].y = gelu_f(ar[j].y);
            ar[j].z = gelu_f(ar[j].z); ar[j].w = gelu_f(ar[j].w);
        }
    }
    v16h afr[4];
#pragma unroll
    for (int s = 0; s < 4; ++s) {
        float4 a0 = ar[s*4+0], a1 = ar[s*4+1], a2 = ar[s*4+2], a3 = ar[s*4+3];
        v16h af;
        af[0]  = (_Float16)a0.x; af[1]  = (_Float16)a0.y; af[2]  = (_Float16)a0.z; af[3]  = (_Float16)a0.w;
        af[4]  = (_Float16)a1.x; af[5]  = (_Float16)a1.y; af[6]  = (_Float16)a1.z; af[7]  = (_Float16)a1.w;
        af[8]  = (_Float16)a2.x; af[9]  = (_Float16)a2.y; af[10] = (_Float16)a2.z; af[11] = (_Float16)a2.w;
        af[12] = (_Float16)a3.x; af[13] = (_Float16)a3.y; af[14] = (_Float16)a3.z; af[15] = (_Float16)a3.w;
        afr[s] = af;
    }

    v8f acc[8] = {};                                // 8 column tiles x 16x16
#pragma unroll
    for (int s = 0; s < 4; ++s) {                   // K = 4 * 32
        int kbase = s * 32 + hlf * 16;              // B-layout: contiguous K run of 16
#pragma unroll
        for (int ct = 0; ct < 8; ++ct) {
            v16h bf = *(const v16h*)&ldsW[(ct * 16 + m) * CC + kbase];
            acc[ct] = __builtin_amdgcn_wmma_f32_16x16x32_f16(
                false, afr[s], false, bf, (short)0, acc[ct], false, false);
        }
    }

    float g = 0.f;
    if (mode == 2) g = 1.f / (1.f + __expf(-skip_ptr[0]));
    int mbase = hlf * 8;                            // C/D layout: VGPR i -> M = i + 8*hlf
#pragma unroll
    for (int ct = 0; ct < 8; ++ct) {
        int n = ct * 16 + m;
        float bv = ldsB[n];
#pragma unroll
        for (int i = 0; i < 8; ++i) {
            int grow = row0 + mbase + i;
            float v = acc[ct][i] + bv;
            if (mode == 1)      v = fmaxf(v, 0.f);
            else if (mode == 2) v = g * v + (1.f - g) * xold[(size_t)grow * CC + n];
            out[(size_t)grow * CC + n] = v;
        }
    }
}

// ---------------- attention edge kernels ----------------------------------
__global__ void edge_logits_kernel(const float* __restrict__ q, const float* __restrict__ kr,
                                   const int* __restrict__ src, const int* __restrict__ dst,
                                   const float* __restrict__ prel,
                                   float* __restrict__ logits, unsigned* __restrict__ maxb) {
    int idx = blockIdx.x * blockDim.x + threadIdx.x;
    if (idx >= EE * HH) return;
    int e = idx >> 3, h = idx & 7;
    int s = src[e], d = dst[e];
    const float4* qp = (const float4*)(q  + (size_t)d * CC + h * DD);
    const float4* kp = (const float4*)(kr + (size_t)s * CC + h * DD);
    float acc = 0.f;
#pragma unroll
    for (int j = 0; j < 4; ++j) {
        float4 a = qp[j], b = kp[j];
        acc += a.x * b.x + a.y * b.y + a.z * b.z + a.w * b.w;
    }
    float lg = acc * prel[h] * 0.25f;               // scale = 1/sqrt(16)
    logits[idx] = lg;
    atomicMax(&maxb[(size_t)d * HH + h], fflip(lg));
}

__global__ void edge_exp_kernel(float* __restrict__ logits, const int* __restrict__ dst,
                                const unsigned* __restrict__ maxb, float* __restrict__ sumb) {
    int idx = blockIdx.x * blockDim.x + threadIdx.x;
    if (idx >= EE * HH) return;
    int e = idx >> 3, h = idx & 7;
    int d = dst[e];
    float m = funflip(maxb[(size_t)d * HH + h]);
    float v = __expf(logits[idx] - m);
    logits[idx] = v;
    atomicAdd(&sumb[(size_t)d * HH + h], v);
}

__global__ void edge_scatter_kernel(const float* __restrict__ ev, const float* __restrict__ sumb,
                                    const float* __restrict__ vr, const int* __restrict__ src,
                                    const int* __restrict__ dst, float* __restrict__ agg) {
    int idx = blockIdx.x * blockDim.x + threadIdx.x;
    if (idx >= EE * HH) return;
    int e = idx >> 3, h = idx & 7;
    int s = src[e], d = dst[e];
    float alpha = ev[idx] / (sumb[(size_t)d * HH + h] + 1e-16f);
    const float4* vp = (const float4*)(vr + (size_t)s * CC + h * DD);
    float* ap = agg + (size_t)d * CC + h * DD;
#pragma unroll
    for (int j = 0; j < 4; ++j) {
        float4 v4 = vp[j];
        atomicAdd(ap + j * 4 + 0, v4.x * alpha);
        atomicAdd(ap + j * 4 + 1, v4.y * alpha);
        atomicAdd(ap + j * 4 + 2, v4.z * alpha);
        atomicAdd(ap + j * 4 + 3, v4.w * alpha);
    }
}

// ---------------- orchestration -------------------------------------------
extern "C" void kernel_launch(void* const* d_in, const int* in_sizes, int n_in,
                              void* d_out, int out_size, void* d_ws, size_t ws_size,
                              hipStream_t stream) {
    (void)in_sizes; (void)n_in; (void)out_size; (void)ws_size;

    const int*   tok_stmt = (const int*)d_in[0];
    const int*   tok_func = (const int*)d_in[1];
    const int*   e_src[3] = {(const int*)d_in[2], (const int*)d_in[4], (const int*)d_in[6]};
    const int*   e_dst[3] = {(const int*)d_in[3], (const int*)d_in[5], (const int*)d_in[7]};
    const float* emb   = (const float*)d_in[8];
    const float* lin_w = (const float*)d_in[9];
    const float* lin_b = (const float*)d_in[10];
    const float* kw    = (const float*)d_in[11];
    const float* kb    = (const float*)d_in[12];
    const float* qw    = (const float*)d_in[13];
    const float* qb    = (const float*)d_in[14];
    const float* vw    = (const float*)d_in[15];
    const float* vb    = (const float*)d_in[16];
    const float* aw    = (const float*)d_in[17];
    const float* ab    = (const float*)d_in[18];
    const float* skip  = (const float*)d_in[19];
    const float* a_rel = (const float*)d_in[20];
    const float* m_rel = (const float*)d_in[21];
    const float* p_rel = (const float*)d_in[22];

    float* X = (float*)d_out;                           // node features live here
    float* Xs[2] = {X, X + (size_t)NS * CC};

    // carve workspace
    float* ws = (float*)d_ws;
    size_t off = 0;
    float*    Q    = ws + off; off += (size_t)(NS + NF) * CC;   // also encoder scratch
    float*    KR   = ws + off; off += (size_t)NS * CC;
    float*    VR   = ws + off; off += (size_t)NS * CC;
    float*    AGG  = ws + off; off += (size_t)(NS + NF) * CC;
    float*    LOGI = ws + off; off += (size_t)EE * HH;
    unsigned* MAXB = (unsigned*)(ws + off); off += (size_t)NS * HH;
    float*    SUMB = ws + off; off += (size_t)NS * HH;
    float*    KWE  = ws + off; off += (size_t)CC * CC;
    float*    KBE  = ws + off; off += CC;
    float*    VWE  = ws + off; off += (size_t)CC * CC;
    float*    VBE  = ws + off; off += CC;

    float* ENC[2]  = {Q, Q + (size_t)NS * CC};          // encoder output (dead after lin)
    float* Qs[2]   = {Q, Q + (size_t)NS * CC};          // reuses ENC storage
    float* AGGs[2] = {AGG, AGG + (size_t)NS * CC};
    const int NN[2] = {NS, NF};

    auto gemm = [&](const float* Ain, const float* Wp, const float* bp, float* Outp,
                    int n, int mode, int agelu, const float* xold, const float* sp) {
        int grid = (n + 127) / 128;                     // 128 rows per block (8 waves)
        gemm_wmma_kernel<<<grid, 256, 0, stream>>>(Ain, Wp, bp, Outp, n, mode, agelu, xold, sp);
    };

    // --- encoder + per-type linear ---
    encoder_kernel<<<NS, CC, 0, stream>>>(tok_stmt, emb, ENC[0]);
    encoder_kernel<<<NF, CC, 0, stream>>>(tok_func, emb, ENC[1]);
    for (int t = 0; t < 2; ++t)
        gemm(ENC[t], lin_w + (size_t)t * CC * CC, lin_b + t * CC, Xs[t], NN[t], 1, 0,
             nullptr, nullptr);

    const int ST[3] = {0, 0, 1}, DT[3] = {0, 1, 0};
    for (int l = 0; l < LL; ++l) {
        // Q projections
        for (int t = 0; t < 2; ++t)
            gemm(Xs[t], qw + ((size_t)l * 2 + t) * CC * CC, qb + ((size_t)l * 2 + t) * CC,
                 Qs[t], NN[t], 0, 0, nullptr, nullptr);
        // zero agg
        {
            long long n = (long long)(NS + NF) * CC;
            fill_u32_kernel<<<(int)((n + 255) / 256), 256, 0, stream>>>((unsigned*)AGG, 0u, n);
        }
        for (int r = 0; r < 3; ++r) {
            int st = ST[r], dt = DT[r];
            make_eff_kernel<<<64, 256, 0, stream>>>(
                kw + ((size_t)l * 2 + st) * CC * CC, kb + ((size_t)l * 2 + st) * CC,
                a_rel + (size_t)(l * 3 + r) * HH * DD * DD, KWE, KBE);
            make_eff_kernel<<<64, 256, 0, stream>>>(
                vw + ((size_t)l * 2 + st) * CC * CC, vb + ((size_t)l * 2 + st) * CC,
                m_rel + (size_t)(l * 3 + r) * HH * DD * DD, VWE, VBE);
            gemm(Xs[st], KWE, KBE, KR, NN[st], 0, 0, nullptr, nullptr); // k_r for all src nodes
            gemm(Xs[st], VWE, VBE, VR, NN[st], 0, 0, nullptr, nullptr); // v_r for all src nodes

            long long nm = (long long)NN[dt] * HH;
            init_softmax_kernel<<<(int)((nm + 255) / 256), 256, 0, stream>>>(MAXB, SUMB, nm);

            long long ne = (long long)EE * HH;
            int gb = (int)((ne + 255) / 256);
            edge_logits_kernel<<<gb, 256, 0, stream>>>(Qs[dt], KR, e_src[r], e_dst[r],
                                                       p_rel + (size_t)(l * 3 + r) * HH, LOGI, MAXB);
            edge_exp_kernel<<<gb, 256, 0, stream>>>(LOGI, e_dst[r], MAXB, SUMB);
            edge_scatter_kernel<<<gb, 256, 0, stream>>>(LOGI, SUMB, VR, e_src[r], e_dst[r], AGGs[dt]);
        }
        // output projection + fused gelu-on-A + gated skip (in place into Xs)
        for (int t = 0; t < 2; ++t)
            gemm(AGGs[t], aw + ((size_t)l * 2 + t) * CC * CC, ab + ((size_t)l * 2 + t) * CC,
                 Xs[t], NN[t], 2, 1, Xs[t], skip + (size_t)l * 2 + t);
    }
}
